// MT_GAT_topk_shareEn_multiple8_joint_last_GCN_70712341561400
// MI455X (gfx1250) — matrix operand order repaired and checked by reading, not compiled, not run on hardware
//
#include <hip/hip_runtime.h>

// ---------------------------------------------------------------------------
// MI455X (gfx1250, wave32) implementation.
// All dense GEMMs run on v_wmma_f32_16x16x32_bf16 (fp32 -> bf16 inputs with
// fp32 accumulation). B operands are kept in BT layout ([N][Kp]) so both A and
// B tile staging are single vectorized global_load_b128 -> ds_store_b128.
// Double-buffered LDS + software pipelining: one barrier per K-step, next
// tile's global loads overlapped with the WMMAs. Out-of-range tile rows are
// handled by pointer clamping (affected outputs are epilogue-masked), so the
// inner loop carries no predication.
// ---------------------------------------------------------------------------

typedef unsigned short u16;
typedef __attribute__((ext_vector_type(16))) __bf16 v16bf;
typedef __attribute__((ext_vector_type(8)))  float  v8f;

#define Bq   128
#define Nn   264
#define Fd   264
#define Fp   288          // 264 padded to mult of 32
#define BN   33792        // Bq*Nn
#define Hd   64
#define DH   132          // head dim (F/2)
#define DHp  160          // 132 padded
#define KP   132          // top-k K
#define HCd  64

__device__ __forceinline__ u16 f2bf(float f) {
  unsigned x = __float_as_uint(f);
  return (u16)((x + 0x7FFFu + ((x >> 16) & 1u)) >> 16);
}
__device__ __forceinline__ float sigmoidf(float x) { return 1.f / (1.f + __expf(-x)); }

union FragBF { v16bf v; unsigned u[8]; };

// ---------------------------------------------------------------------------
// Generic bf16 WMMA GEMM: C = act(A[M x Kp] * B + bias), B given as BT [N x Kp]
// ---------------------------------------------------------------------------
#define AS_LD 40
#define BS_LD 40

template <int ACT>  // 0=none 1=relu 2=sigmoid
__global__ __launch_bounds__(256) void k_gemm_bf16(
    const u16* __restrict__ A, const u16* __restrict__ B,
    const float* __restrict__ bias, float* __restrict__ C, u16* __restrict__ Cb,
    int M, int N, int Kp, int ldc, int ldcb,
    long sA, long sB, long sC, long sCb)
{
  __shared__ u16 As[2][64 * AS_LD];
  __shared__ u16 Bs[2][64 * BS_LD];

  const int bz = blockIdx.z;
  A += (long)bz * sA;
  B += (long)bz * sB;
  if (C)  C  += (long)bz * sC;
  if (Cb) Cb += (long)bz * sCb;

  const int m0 = blockIdx.y * 64, n0 = blockIdx.x * 64;
  const int tid  = threadIdx.x;
  const int lane = tid & 31, wid = tid >> 5;
  const int wm = (wid & 3) * 16;   // wave row strip
  const int wn = (wid >> 2) * 32;  // wave col strip (2 wmma tiles)

  v8f acc0 = {}; v8f acc1 = {};

  const int sr = tid >> 2;        // 0..63 staging row (A: m, B: n)
  const int kc = (tid & 3) * 8;   // 0..24 staging k-group (8 bf16 = 16B)
  const int gm = m0 + sr;
  const int gn = n0 + sr;
  // Clamp out-of-range staging rows instead of predicating: garbage rows only
  // feed C rows >= M / C cols >= N, which the epilogue masks out.
  const u16* __restrict__ Arow = A + (long)((gm < M) ? gm : (M - 1)) * Kp + kc;
  const u16* __restrict__ Brow = B + (long)((gn < N) ? gn : (N - 1)) * Kp + kc;

  // ---- software pipeline prologue: tile 0 in registers
  uint4 ra = *(const uint4*)(Arow);
  uint4 rb = *(const uint4*)(Brow);

  int buf = 0;
  for (int k0 = 0; k0 < Kp; k0 += 32, buf ^= 1) {
    // commit staged registers to LDS (double buffer)
    *(uint4*)(&As[buf][sr * AS_LD + kc]) = ra;
    *(uint4*)(&Bs[buf][sr * BS_LD + kc]) = rb;
    __syncthreads();

    // issue next tile's global loads; latency hidden behind the WMMAs below
    if (k0 + 32 < Kp) {
      ra = *(const uint4*)(Arow + k0 + 32);
      rb = *(const uint4*)(Brow + k0 + 32);
      __builtin_prefetch((const void*)(Arow + k0 + 64), 0, 1);  // speculative
      __builtin_prefetch((const void*)(Brow + k0 + 64), 0, 1);
    }

    // ---- build fragments (contiguous k order, k = 16*(lane/16) + s, same for A and B)
    FragBF af, bf0, bf1;
    const int am = wm + (lane & 15);
    const int hk = (lane & 16) ? 16 : 0;
    #pragma unroll
    for (int j = 0; j < 8; ++j) {
      const int kk = hk + j * 2;  // A: 16x32 MxK
      af.u[j] = *(const unsigned*)(&As[buf][am * AS_LD + kk]);
    }
    const int bn0 = wn + (lane & 15);
    const int bn1 = bn0 + 16;
    const int kb  = (lane & 16) ? 16 : 0;                    // B: 32x16 KxN
    #pragma unroll
    for (int j = 0; j < 8; ++j) {
      bf0.u[j] = *(const unsigned*)(&Bs[buf][bn0 * BS_LD + kb + j * 2]);
      bf1.u[j] = *(const unsigned*)(&Bs[buf][bn1 * BS_LD + kb + j * 2]);
    }

    acc0 = __builtin_amdgcn_wmma_f32_16x16x32_bf16(false, af.v, false, bf0.v,
                                                   (short)0, acc0, false, false);
    acc1 = __builtin_amdgcn_wmma_f32_16x16x32_bf16(false, af.v, false, bf1.v,
                                                   (short)0, acc1, false, false);
  }

  // ---- epilogue (C/D layout: vgpr v -> row v (+8 for hi lanes), col = lane&15)
  const int rbase = m0 + wm + ((lane & 16) ? 8 : 0);
  const int c0 = n0 + wn + (lane & 15);
  const int c1 = c0 + 16;
  #pragma unroll
  for (int v = 0; v < 8; ++v) {
    const int gr = rbase + v;
    if (gr >= M) continue;
    float f0 = acc0[v], f1 = acc1[v];
    if (bias) { if (c0 < N) f0 += bias[c0]; if (c1 < N) f1 += bias[c1]; }
    if (ACT == 1) { f0 = fmaxf(f0, 0.f); f1 = fmaxf(f1, 0.f); }
    if (ACT == 2) { f0 = sigmoidf(f0); f1 = sigmoidf(f1); }
    if (C) {
      if (c0 < N) C[(long)gr * ldc + c0] = f0;
      if (c1 < N) C[(long)gr * ldc + c1] = f1;
    }
    if (Cb) {
      if (c0 < ldcb) Cb[(long)gr * ldcb + c0] = (c0 < N) ? f2bf(f0) : (u16)0;
      if (c1 < ldcb) Cb[(long)gr * ldcb + c1] = (c1 < N) ? f2bf(f1) : (u16)0;
    }
  }
}

// ---------------------------------------------------------------------------
// Data movement / conversion kernels
// ---------------------------------------------------------------------------
__global__ void k_zero_b(u16* p, long n) {
  long i = (long)blockIdx.x * 256 + threadIdx.x; if (i < n) p[i] = 0;
}
__global__ void k_copy_f(const float* a, float* b, long n) {
  long i = (long)blockIdx.x * 256 + threadIdx.x; if (i < n) b[i] = a[i];
}
// weight [bat][K x N] -> bf16 BT [bat][N x Kp] (transpose + zero-pad k)
__global__ void k_cvt_bt(const float* src, u16* dst, int K, int N, int Kp, int bat) {
  long i = (long)blockIdx.x * 256 + threadIdx.x;
  long tot = (long)bat * N * Kp; if (i >= tot) return;
  int k = (int)(i % Kp); long rn = i / Kp;
  int n = (int)(rn % N); int b = (int)(rn / N);
  dst[i] = (k < K) ? f2bf(src[((long)b * K + k) * N + n]) : (u16)0;
}
// weight already [N x K] (torch Linear) -> bf16 BT [N x Kp] (pad k only)
__global__ void k_cvt_pad(const float* src, u16* dst, int Nrows, int K, int Kp) {
  long i = (long)blockIdx.x * 256 + threadIdx.x;
  long tot = (long)Nrows * Kp; if (i >= tot) return;
  int k = (int)(i % Kp); long n = i / Kp;
  dst[i] = (k < K) ? f2bf(src[n * K + k]) : (u16)0;
}
// activation [rows x C] -> bf16 [rows x Cb] (zero-pad cols), optional relu
__global__ void k_cvt_act(const float* src, u16* dst, long rows, int C, int Cb, int relu) {
  long i = (long)blockIdx.x * 256 + threadIdx.x;
  if (i >= rows * (long)Cb) return;
  int c = (int)(i % Cb); long r = i / Cb;
  if (c < C) {
    float v = src[r * C + c];
    if (relu) v = fmaxf(v, 0.f);
    dst[i] = f2bf(v);
  } else dst[i] = 0;
}

// ---------------------------------------------------------------------------
// GINE edge scatter: h[dst] += relu(x[src] + ea*We + be)
// ---------------------------------------------------------------------------
__global__ __launch_bounds__(256) void k_gine_edge(
    const int* __restrict__ src, const int* __restrict__ dst,
    const float* __restrict__ ew, const float* __restrict__ We,
    const float* __restrict__ be, const float* __restrict__ x,
    float* __restrict__ h, int E)
{
  int e = blockIdx.x; if (e >= E) return;
  int s = src[e], d = dst[e];
  float w = ew[e];
  for (int f = threadIdx.x; f < Fd; f += 256) {
    float m = fmaxf(x[(long)s * Fd + f] + w * We[f] + be[f], 0.f);
    atomicAdd(&h[(long)d * Fd + f], m);
  }
}

// BatchNorm(a+b) fused, optional relu + bf16 padded copy
__global__ void k_bn2(const float* a, const float* b, const float* g, const float* bet,
                      float* outf, u16* outb, long rows, int C, int Cb, float s, int relu) {
  long i = (long)blockIdx.x * 256 + threadIdx.x;
  if (i >= rows * (long)Cb) return;
  int c = (int)(i % Cb); long r = i / Cb;
  if (c < C) {
    float v = g[c] * ((a[r * C + c] + b[r * C + c]) * s) + bet[c];
    if (relu) v = fmaxf(v, 0.f);
    if (outf) outf[r * C + c] = v;
    if (outb) outb[i] = f2bf(v);
  } else if (outb) outb[i] = 0;
}
// out = a + b (fp32 + padded bf16)
__global__ void k_add_bf(const float* a, const float* b, float* outf, u16* outb,
                         long rows, int C, int Cb) {
  long i = (long)blockIdx.x * 256 + threadIdx.x;
  if (i >= rows * (long)Cb) return;
  int c = (int)(i % Cb); long r = i / Cb;
  if (c < C) {
    float v = a[r * C + c] + b[r * C + c];
    outf[r * C + c] = v; outb[i] = f2bf(v);
  } else outb[i] = 0;
}

// ---------------------------------------------------------------------------
// Attention helpers
// ---------------------------------------------------------------------------
// q -> [bh][264 x 160] (A operand), k -> [bh][264 x 160] (BT operand: rows=key n),
// v -> [bh][132 x 288] (BT operand: rows=head-dim c, cols=node)
__global__ void k_repack_qkv(const float* __restrict__ qkv, u16* qb, u16* kbt, u16* vbt) {
  long i = (long)blockIdx.x * 256 + threadIdx.x;
  if (i >= (long)BN * Fd) return;
  int node = (int)(i / Fd), f = (int)(i % Fd);
  int b = node / Nn, n = node % Nn;
  int hd = f / DH, c = f % DH;
  int bh = b * 2 + hd;
  float qv = qkv[(long)node * 792 + f];
  float kv = qkv[(long)node * 792 + 264 + f];
  float vv = qkv[(long)node * 792 + 528 + f];
  qb [(long)bh * (Nn * DHp) + (long)n * DHp + c] = f2bf(qv);
  kbt[(long)bh * (Nn * DHp) + (long)n * DHp + c] = f2bf(kv);
  vbt[(long)bh * (DH * Fp)  + (long)c * Fp  + n] = f2bf(vv);
}

__global__ __launch_bounds__(256) void k_softmax(const float* __restrict__ S,
                                                 u16* __restrict__ P, float scale) {
  __shared__ float red[256];
  const int row = blockIdx.x, bh = blockIdx.y, t = threadIdx.x;
  const float* s = S + ((long)bh * Nn + row) * Nn;
  const int c2 = t + 256;
  float v1 = s[t] * scale;
  float v2 = (c2 < Nn) ? s[c2] * scale : -3.0e38f;
  red[t] = fmaxf(v1, v2); __syncthreads();
  for (int o = 128; o > 0; o >>= 1) { if (t < o) red[t] = fmaxf(red[t], red[t + o]); __syncthreads(); }
  const float mx = red[0]; __syncthreads();
  float e1 = __expf(v1 - mx);
  float e2 = (c2 < Nn) ? __expf(v2 - mx) : 0.f;
  red[t] = e1 + e2; __syncthreads();
  for (int o = 128; o > 0; o >>= 1) { if (t < o) red[t] += red[t + o]; __syncthreads(); }
  const float inv = 1.f / red[0];
  u16* p = P + ((long)bh * Nn + row) * Fp;
  p[t] = f2bf(e1 * inv);
  if (c2 < Fp) p[c2] = (c2 < Nn) ? f2bf(e2 * inv) : (u16)0;
}

__global__ void k_repack_o(const float* __restrict__ o, u16* __restrict__ ob) {
  long i = (long)blockIdx.x * 256 + threadIdx.x;
  if (i >= (long)BN * Fp) return;
  int f = (int)(i % Fp); long node = i / Fp;
  if (f >= Fd) { ob[i] = 0; return; }
  int b = (int)(node / Nn), n = (int)(node % Nn);
  int hd = f / DH, c = f % DH;
  ob[i] = f2bf(o[((long)(b * 2 + hd) * Nn + n) * DH + c]);
}

// ---------------------------------------------------------------------------
// GCN
// ---------------------------------------------------------------------------
__global__ void k_deg_init(float* deg, int n) {
  int i = blockIdx.x * 256 + threadIdx.x; if (i < n) deg[i] = 1.f;
}
__global__ void k_deg_acc(const int* dst, const float* ew, float* deg, int E) {
  int e = blockIdx.x * 256 + threadIdx.x; if (e < E) atomicAdd(&deg[dst[e]], ew[e]);
}
__global__ void k_dinv(float* deg, int n) {
  int i = blockIdx.x * 256 + threadIdx.x; if (i < n) deg[i] = rsqrtf(deg[i]);
}
__global__ void k_gcn_self(const float* xw, const float* dinv, const float* b, float* out) {
  long i = (long)blockIdx.x * 256 + threadIdx.x;
  if (i >= (long)BN * Hd) return;
  int n = (int)(i / Hd), h = (int)(i % Hd);
  out[i] = xw[i] * dinv[n] * dinv[n] + b[h];
}
__global__ __launch_bounds__(64) void k_gcn_edge(const int* src, const int* dst,
    const float* ew, const float* dinv, const float* xw, float* out, int E) {
  int e = blockIdx.x; if (e >= E) return;
  int s = src[e], d = dst[e];
  float c = dinv[s] * ew[e] * dinv[d];
  int t = threadIdx.x;
  atomicAdd(&out[(long)d * Hd + t], xw[(long)s * Hd + t] * c);
}
__global__ void k_relu_bf(const float* in, float* outf, u16* outb, long n) {
  long i = (long)blockIdx.x * 256 + threadIdx.x; if (i >= n) return;
  float v = fmaxf(in[i], 0.f); outf[i] = v; outb[i] = f2bf(v);
}

// ---------------------------------------------------------------------------
// Small reductions / head logic
// ---------------------------------------------------------------------------
__global__ __launch_bounds__(64) void k_gemv(const float* A, const float* w,
    const float* bias, float* out, int K, int relu_in) {
  __shared__ float red[64];
  const int r = blockIdx.x, t = threadIdx.x;
  float acc = 0.f;
  for (int k = t; k < K; k += 64) {
    float a = A[(long)r * K + k];
    if (relu_in) a = fmaxf(a, 0.f);
    acc += a * w[k];
  }
  red[t] = acc; __syncthreads();
  for (int o = 32; o > 0; o >>= 1) { if (t < o) red[t] += red[t + o]; __syncthreads(); }
  if (t == 0) out[r] = red[0] + (bias ? bias[0] : 0.f);
}

__global__ void k_a4(const float* x11, const float* x22, const float* w4,
                     const float* b4, float* a) {
  int node = blockIdx.x * 256 + threadIdx.x; if (node >= BN) return;
  float s = b4[0];
  for (int h = 0; h < Hd; ++h)
    s += x11[(long)node * Hd + h] * w4[h] + x22[(long)node * Hd + h] * w4[Hd + h];
  a[node] = s;
}

__global__ void k_wpnorm(const float* wp, float* nrm) {
  int i = threadIdx.x;
  if (i < 8) {
    float s = 0.f;
    for (int k = 0; k < Hd; ++k) s += wp[i * Hd + k] * wp[i * Hd + k];
    nrm[i] = sqrtf(s);
  }
}

__global__ void k_base_score(const float* x11, const float* x22, const float* att,
                             const float* wp, const float* nrm, int which,
                             float* base, float* score) {
  int node = blockIdx.x * 256 + threadIdx.x; if (node >= BN) return;
  float at = att[node], acc = 0.f;
  for (int h = 0; h < Hd; ++h) {
    float v = which ? (x22[(long)node * Hd + h] + at * x11[(long)node * Hd + h])
                    : (x11[(long)node * Hd + h] + at * x22[(long)node * Hd + h]);
    base[(long)node * Hd + h] = v;
    acc += v * wp[h];
  }
  score[node] = sigmoidf(acc / nrm[0]);
}

// top-K=132 of 264 per graph (descending, lowest-index tie break == lax.top_k)
__global__ __launch_bounds__(288) void k_topk(const float* __restrict__ score,
    const float* __restrict__ base, u16* __restrict__ xpb,
    float* __restrict__ out_perm, float* __restrict__ out_sc) {
  __shared__ float s[Nn];
  const int b = blockIdx.x, t = threadIdx.x;
  if (t < Nn) s[t] = score[(long)b * Nn + t];
  __syncthreads();
  if (t >= Nn) return;
  const float mine = s[t];
  int rank = 0;
  for (int j = 0; j < Nn; ++j) {
    float o = s[j];
    rank += (o > mine) || (o == mine && j < t);
  }
  if (rank < KP) {
    const int node = b * Nn + t;
    const long pos = (long)b * KP + rank;
    out_perm[pos] = (float)node;
    out_sc[pos]   = sigmoidf(mine);  // reference double-sigmoids the vals
    for (int h = 0; h < Hd; ++h)
      xpb[pos * Hd + h] = f2bf(base[(long)node * Hd + h] * mine);
  }
}

// ---------------------------------------------------------------------------
// Host launcher
// ---------------------------------------------------------------------------
static inline dim3 g1(long n) { return dim3((unsigned)((n + 255) / 256)); }

static inline void gemm(hipStream_t st, int act, const u16* A, const u16* B,
                        const float* bias, float* C, u16* Cb,
                        int M, int N, int Kp, int ldc, int ldcb,
                        int batch = 1, long sA = 0, long sB = 0, long sC = 0, long sCb = 0) {
  dim3 g((N + 63) / 64, (M + 63) / 64, batch), blk(256);
  if (act == 0)      k_gemm_bf16<0><<<g, blk, 0, st>>>(A, B, bias, C, Cb, M, N, Kp, ldc, ldcb, sA, sB, sC, sCb);
  else if (act == 1) k_gemm_bf16<1><<<g, blk, 0, st>>>(A, B, bias, C, Cb, M, N, Kp, ldc, ldcb, sA, sB, sC, sCb);
  else               k_gemm_bf16<2><<<g, blk, 0, st>>>(A, B, bias, C, Cb, M, N, Kp, ldc, ldcb, sA, sB, sC, sCb);
}

extern "C" void kernel_launch(void* const* d_in, const int* in_sizes, int n_in,
                              void* d_out, int out_size, void* d_ws, size_t ws_size,
                              hipStream_t stream) {
  // ---- inputs (setup_inputs order, params dict flattened in insertion order)
  const float* x   = (const float*)d_in[0];
  const int*   ei  = (const int*)d_in[1];
  const float* ew  = (const float*)d_in[2];
  const int E = in_sizes[1] / 2;
  const int* esrc = ei;
  const int* edst = ei + E;
  int p = 4;
  const float *We = (const float*)d_in[p++], *be = (const float*)d_in[p++];
  const float *W1 = (const float*)d_in[p++], *b1 = (const float*)d_in[p++];
  const float *W2 = (const float*)d_in[p++], *b2 = (const float*)d_in[p++];
  const float *Win = (const float*)d_in[p++], *bin = (const float*)d_in[p++];
  const float *Wout = (const float*)d_in[p++], *bout = (const float*)d_in[p++];
  const float *n1g = (const float*)d_in[p++], *n1b = (const float*)d_in[p++];
  const float *n2g = (const float*)d_in[p++], *n2b = (const float*)d_in[p++];
  const float *n3g = (const float*)d_in[p++], *n3b = (const float*)d_in[p++];
  const float *Wm1 = (const float*)d_in[p++], *bm1 = (const float*)d_in[p++];
  const float *Wm2 = (const float*)d_in[p++], *bm2 = (const float*)d_in[p++];
  const float *Wg1 = (const float*)d_in[p++], *bg1 = (const float*)d_in[p++];
  const float *Wg2 = (const float*)d_in[p++], *bg2 = (const float*)d_in[p++];
  const float *W111 = (const float*)d_in[p++], *b111 = (const float*)d_in[p++];
  const float *W1111 = (const float*)d_in[p++], *b1111 = (const float*)d_in[p++];
  const float *W222 = (const float*)d_in[p++], *b222 = (const float*)d_in[p++];
  const float *W2222 = (const float*)d_in[p++], *b2222 = (const float*)d_in[p++];
  const float *w4 = (const float*)d_in[p++], *b4 = (const float*)d_in[p++];
  const float *W4l = (const float*)d_in[p++], *b4l = (const float*)d_in[p++];
  const float *W5l = (const float*)d_in[p++], *b5l = (const float*)d_in[p++];
  const float *wpv = (const float*)d_in[p++];
  const float *WA = (const float*)d_in[p++], *bA = (const float*)d_in[p++];
  const float *WB = (const float*)d_in[p++], *bB = (const float*)d_in[p++];
  float* out = (float*)d_out;

  // ---- workspace bump allocator
  char* base_ws = (char*)d_ws;
  size_t off = 0;
  auto alloc = [&](size_t bytes) -> void* {
    void* q = base_ws + off;
    off = (off + bytes + 255) & ~(size_t)255;
    return q;
  };
  // bf16 weights, all in BT layout [N][Kp]
  u16* w1b   = (u16*)alloc((size_t)Fd * Fp * 2);           // [264][288]
  u16* w2b   = (u16*)alloc((size_t)Fd * Fp * 2);
  u16* winb  = (u16*)alloc((size_t)792 * Fp * 2);          // [792][288]
  u16* woutb = (u16*)alloc((size_t)Fd * Fp * 2);
  u16* wm1b  = (u16*)alloc((size_t)528 * Fp * 2);          // [528][288]
  u16* wm2b  = (u16*)alloc((size_t)Fd * 544 * 2);          // [264][544]
  u16* wg1b  = (u16*)alloc((size_t)Hd * Fp * 2);           // [64][288]
  u16* wg2b  = (u16*)alloc((size_t)Hd * Fp * 2);
  u16* w111b = (u16*)alloc((size_t)Hd * Nn * Hd * 2);      // [64][16896]
  u16* w222b = (u16*)alloc((size_t)Hd * Nn * Hd * 2);
  u16* w4lb  = (u16*)alloc((size_t)66 * Fp * 2);           // [66][288]
  u16* w5lb  = (u16*)alloc((size_t)Nn * 96 * 2);           // [264][96]
  u16* wab   = (u16*)alloc((size_t)8 * HCd * KP * Hd * 2); // 8 x [64][8448]
  // activations
  u16*   xbf   = (u16*)alloc((size_t)BN * Fp * 2);
  float* h     = (float*)alloc((size_t)BN * Fd * 4);
  u16*   hbf   = (u16*)alloc((size_t)BN * Fp * 2);
  u16*   t1bf  = (u16*)alloc((size_t)BN * Fp * 2);
  float* gine  = (float*)alloc((size_t)BN * Fd * 4);
  float* hloc  = (float*)alloc((size_t)BN * Fd * 4);
  float* qkv   = (float*)alloc((size_t)BN * 792 * 4);
  u16*   qb    = (u16*)alloc((size_t)256 * Nn * DHp * 2);
  u16*   kbt   = (u16*)alloc((size_t)256 * Nn * DHp * 2);
  u16*   vbt   = (u16*)alloc((size_t)256 * DH * Fp * 2);
  float* sc    = (float*)alloc((size_t)256 * Nn * Nn * 4);
  u16*   prb   = (u16*)alloc((size_t)256 * Nn * Fp * 2);
  float* ofp   = (float*)alloc((size_t)256 * Nn * DH * 4);
  u16*   obf   = (u16*)alloc((size_t)BN * Fp * 2);
  float* mha   = (float*)alloc((size_t)BN * Fd * 4);
  float* hatt  = (float*)alloc((size_t)BN * Fd * 4);
  float* out0  = (float*)alloc((size_t)BN * Fd * 4);
  u16*   out0b = (u16*)alloc((size_t)BN * Fp * 2);
  u16*   t2b   = (u16*)alloc((size_t)BN * 544 * 2);
  float* mlp   = (float*)alloc((size_t)BN * Fd * 4);
  float* gps   = (float*)alloc((size_t)BN * Fd * 4);
  u16*   gpsb  = (u16*)alloc((size_t)BN * Fp * 2);
  float* xw1   = (float*)alloc((size_t)BN * Hd * 4);
  float* xw2   = (float*)alloc((size_t)BN * Hd * 4);
  float* deg   = (float*)alloc((size_t)BN * 4);
  float* g1a   = (float*)alloc((size_t)BN * Hd * 4);
  float* g2a   = (float*)alloc((size_t)BN * Hd * 4);
  float* x11   = (float*)alloc((size_t)BN * Hd * 4);
  float* x22   = (float*)alloc((size_t)BN * Hd * 4);
  u16*   x11b  = (u16*)alloc((size_t)BN * Hd * 2);
  u16*   x22b  = (u16*)alloc((size_t)BN * Hd * 2);
  float* x111t = (float*)alloc((size_t)Bq * HCd * 4);
  float* x222t = (float*)alloc((size_t)Bq * HCd * 4);
  float* avec  = (float*)alloc((size_t)BN * 4);
  u16*   abf   = (u16*)alloc((size_t)Bq * Fp * 2);
  u16*   a2b   = (u16*)alloc((size_t)Bq * 96 * 2);
  float* att   = (float*)alloc((size_t)BN * 4);
  float* wpn   = (float*)alloc(8 * 4);
  float* basef = (float*)alloc((size_t)BN * Hd * 4);
  float* scor  = (float*)alloc((size_t)BN * 4);
  u16*   xpb   = (u16*)alloc((size_t)Bq * KP * Hd * 2);
  float* hh    = (float*)alloc((size_t)Bq * HCd * 4);
  (void)ws_size; (void)n_in; (void)out_size;

  const float bns = 1.0f / sqrtf(1.0f + 1e-5f);

  // ---- weight prep (bf16 BT layout, K-padded)
  k_cvt_bt <<<g1((long)Fd * Fp), 256, 0, stream>>>(W1, w1b, Fd, Fd, Fp, 1);
  k_cvt_bt <<<g1((long)Fd * Fp), 256, 0, stream>>>(W2, w2b, Fd, Fd, Fp, 1);
  k_cvt_pad<<<g1((long)792 * Fp), 256, 0, stream>>>(Win, winb, 792, Fd, Fp);
  k_cvt_pad<<<g1((long)Fd * Fp), 256, 0, stream>>>(Wout, woutb, Fd, Fd, Fp);
  k_cvt_bt <<<g1((long)528 * Fp), 256, 0, stream>>>(Wm1, wm1b, Fd, 528, Fp, 1);
  k_cvt_bt <<<g1((long)Fd * 544), 256, 0, stream>>>(Wm2, wm2b, 528, Fd, 544, 1);
  k_cvt_bt <<<g1((long)Hd * Fp), 256, 0, stream>>>(Wg1, wg1b, Fd, Hd, Fp, 1);
  k_cvt_bt <<<g1((long)Hd * Fp), 256, 0, stream>>>(Wg2, wg2b, Fd, Hd, Fp, 1);
  k_cvt_bt <<<g1((long)Hd * Nn * Hd), 256, 0, stream>>>(W111, w111b, Nn * Hd, HCd, Nn * Hd, 1);
  k_cvt_bt <<<g1((long)Hd * Nn * Hd), 256, 0, stream>>>(W222, w222b, Nn * Hd, HCd, Nn * Hd, 1);
  k_cvt_bt <<<g1((long)66 * Fp), 256, 0, stream>>>(W4l, w4lb, Nn, 66, Fp, 1);
  k_cvt_bt <<<g1((long)Nn * 96), 256, 0, stream>>>(W5l, w5lb, 66, Nn, 96, 1);
  k_cvt_bt <<<g1((long)8 * HCd * KP * Hd), 256, 0, stream>>>(WA, wab, KP * Hd, HCd, KP * Hd, 8);

  // ---- GPS: GINE branch
  k_cvt_act<<<g1((long)BN * Fp), 256, 0, stream>>>(x, xbf, BN, Fd, Fp, 0);
  k_copy_f<<<g1((long)BN * Fd), 256, 0, stream>>>(x, h, (long)BN * Fd);
  k_gine_edge<<<dim3(E), 256, 0, stream>>>(esrc, edst, ew, We, be, x, h, E);
  k_cvt_act<<<g1((long)BN * Fp), 256, 0, stream>>>(h, hbf, BN, Fd, Fp, 0);
  gemm(stream, 1, hbf, w1b, b1, nullptr, t1bf, BN, Fd, Fp, 0, Fp);
  gemm(stream, 0, t1bf, w2b, b2, gine, nullptr, BN, Fd, Fp, Fd, 0);
  k_bn2<<<g1((long)BN * Fd), 256, 0, stream>>>(gine, x, n1g, n1b, hloc, nullptr, BN, Fd, Fd, bns, 0);

  // ---- GPS: attention branch
  gemm(stream, 0, xbf, winb, bin, qkv, nullptr, BN, 792, Fp, 792, 0);
  k_zero_b<<<g1((long)256 * Nn * DHp), 256, 0, stream>>>(qb, (long)256 * Nn * DHp);
  k_zero_b<<<g1((long)256 * Nn * DHp), 256, 0, stream>>>(kbt, (long)256 * Nn * DHp);
  k_zero_b<<<g1((long)256 * DH * Fp), 256, 0, stream>>>(vbt, (long)256 * DH * Fp);
  k_repack_qkv<<<g1((long)BN * Fd), 256, 0, stream>>>(qkv, qb, kbt, vbt);
  gemm(stream, 0, qb, kbt, nullptr, sc, nullptr, Nn, Nn, DHp, Nn, 0,
       256, (long)Nn * DHp, (long)Nn * DHp, (long)Nn * Nn, 0);
  k_softmax<<<dim3(Nn, 256), 256, 0, stream>>>(sc, prb, rsqrtf((float)DH));
  gemm(stream, 0, prb, vbt, nullptr, ofp, nullptr, Nn, DH, Fp, DH, 0,
       256, (long)Nn * Fp, (long)DH * Fp, (long)Nn * DH, 0);
  k_repack_o<<<g1((long)BN * Fp), 256, 0, stream>>>(ofp, obf);
  gemm(stream, 0, obf, woutb, bout, mha, nullptr, BN, Fd, Fp, Fd, 0);
  k_bn2<<<g1((long)BN * Fd), 256, 0, stream>>>(mha, x, n2g, n2b, hatt, nullptr, BN, Fd, Fd, bns, 0);

  // ---- GPS: combine + MLP + final BN/relu
  k_add_bf<<<g1((long)BN * Fp), 256, 0, stream>>>(hloc, hatt, out0, out0b, BN, Fd, Fp);
  gemm(stream, 1, out0b, wm1b, bm1, nullptr, t2b, BN, 528, Fp, 0, 544);
  gemm(stream, 0, t2b, wm2b, bm2, mlp, nullptr, BN, Fd, 544, Fd, 0);
  k_bn2<<<g1((long)BN * Fp), 256, 0, stream>>>(out0, mlp, n3g, n3b, gps, gpsb, BN, Fd, Fp, bns, 1);

  // ---- GCN x2
  gemm(stream, 0, gpsb, wg1b, nullptr, xw1, nullptr, BN, Hd, Fp, Hd, 0);
  gemm(stream, 0, gpsb, wg2b, nullptr, xw2, nullptr, BN, Hd, Fp, Hd, 0);
  k_deg_init<<<g1(BN), 256, 0, stream>>>(deg, BN);
  k_deg_acc<<<g1(E), 256, 0, stream>>>(edst, ew, deg, E);
  k_dinv<<<g1(BN), 256, 0, stream>>>(deg, BN);
  k_gcn_self<<<g1((long)BN * Hd), 256, 0, stream>>>(xw1, deg, bg1, g1a);
  k_gcn_edge<<<dim3(E), 64, 0, stream>>>(esrc, edst, ew, deg, xw1, g1a, E);
  k_relu_bf<<<g1((long)BN * Hd), 256, 0, stream>>>(g1a, x11, x11b, (long)BN * Hd);
  k_gcn_self<<<g1((long)BN * Hd), 256, 0, stream>>>(xw2, deg, bg2, g2a);
  k_gcn_edge<<<dim3(E), 64, 0, stream>>>(esrc, edst, ew, deg, xw2, g2a, E);
  k_relu_bf<<<g1((long)BN * Hd), 256, 0, stream>>>(g2a, x22, x22b, (long)BN * Hd);

  // ---- joint heads x111/x222 ([128 x 16896] GEMMs)
  const long OUT_HS = 128 + Bq * KP + Bq * KP;        // 33920 per pooled head
  gemm(stream, 0, x11b, w111b, b111, x111t, nullptr, Bq, HCd, Nn * Hd, HCd, 0);
  k_gemv<<<dim3(Bq), 64, 0, stream>>>(x111t, W1111, b1111, out + 8 * OUT_HS + BN, HCd, 1);
  gemm(stream, 0, x22b, w222b, b222, x222t, nullptr, Bq, HCd, Nn * Hd, HCd, 0);
  k_gemv<<<dim3(Bq), 64, 0, stream>>>(x222t, W2222, b2222, out + 8 * OUT_HS + BN + Bq, HCd, 1);

  // ---- node attention gate
  k_a4<<<g1(BN), 256, 0, stream>>>(x11, x22, w4, b4, avec);
  k_cvt_act<<<g1((long)Bq * Fp), 256, 0, stream>>>(avec, abf, Bq, Nn, Fp, 0);
  gemm(stream, 1, abf, w4lb, b4l, nullptr, a2b, Bq, 66, Fp, 0, 96);
  gemm(stream, 2, a2b, w5lb, b5l, att, nullptr, Bq, Nn, 96, Nn, 0);
  k_copy_f<<<g1(BN), 256, 0, stream>>>(att, out + 8 * OUT_HS, BN);

  // ---- 8 pooled heads: score -> topK -> gather -> GEMM -> GEMV
  k_wpnorm<<<dim3(1), 64, 0, stream>>>(wpv, wpn);
  for (int i = 0; i < 8; ++i) {
    k_base_score<<<g1(BN), 256, 0, stream>>>(x11, x22, att, wpv + i * Hd, wpn + i,
                                             (i < 4) ? 0 : 1, basef, scor);
    k_topk<<<dim3(Bq), 288, 0, stream>>>(scor, basef, xpb,
                                         out + i * OUT_HS + Bq,
                                         out + i * OUT_HS + Bq + (long)Bq * KP);
    gemm(stream, 1, xpb, wab + (size_t)i * HCd * KP * Hd, bA + i * HCd, hh, nullptr,
         Bq, HCd, KP * Hd, HCd, 0);
    k_gemv<<<dim3(Bq), 64, 0, stream>>>(hh, WB + i * HCd, bB + i, out + i * OUT_HS, HCd, 0);
  }
}
